// Model_59382217835041
// MI455X (gfx1250) — matrix operand order repaired and checked by reading, not compiled, not run on hardware
//
#include <hip/hip_runtime.h>

// ---------------------------------------------------------------------------
// Equivariant message-passing layer for MI455X (gfx1250, wave32, WMMA).
// Edge MLP on v_wmma_f32_16x16x32_bf16; segment_sum via global_atomic_add_f32
// into an L2-resident accumulator in d_ws. Vectorized LDS traffic (b128),
// branch-free A-fragment construction, cheap SiLU (v_exp + v_rcp).
// ---------------------------------------------------------------------------

typedef __attribute__((ext_vector_type(16))) __bf16        v16bf;
typedef __attribute__((ext_vector_type(8)))  float         v8f;
typedef __attribute__((ext_vector_type(4)))  float         v4f;
typedef __attribute__((ext_vector_type(4)))  unsigned int  v4u;

#define NB_   8      // bessel basis
#define H_    64     // hidden
#define NSH_  9      // spherical harmonics
#define C_    16     // channels
#define OC_   144    // NSH_ * C_
#define SHP_  12     // padded SH row (16B-aligned rows)
#define RCUT_ 5.0f
#define PI_   3.14159265358979f

__device__ __forceinline__ unsigned short f2bf(float f) {
  unsigned int u = __float_as_uint(f);
  u += 0x7FFFu + ((u >> 16) & 1u);       // round-to-nearest-even
  return (unsigned short)(u >> 16);
}
__device__ __forceinline__ unsigned int pkbf(float lo, float hi) {
  return (unsigned int)f2bf(lo) | ((unsigned int)f2bf(hi) << 16);
}

union AFrag { v16bf v; v4u q[2]; };

// ---------------------------------------------------------------------------
// Pre-pack weights into per-lane B-operand (16-bit, Kx16) fragment layout:
// lane = g*16 + n holds K = g*16 + e  (e = element 0..15), column = tile*16+n
// ---------------------------------------------------------------------------
__global__ void prep_frags(const float* __restrict__ w1,
                           const float* __restrict__ w2,
                           unsigned short* __restrict__ w1f,
                           unsigned short* __restrict__ w2f) {
  int tid    = blockIdx.x * blockDim.x + threadIdx.x;
  int stride = gridDim.x * blockDim.x;
  // layer 1: 4 N-tiles (64 cols), K padded 8 -> 32
  for (int i = tid; i < 4 * 32 * 16; i += stride) {
    int e = i & 15, lane = (i >> 4) & 31, t = i >> 9;
    int g = lane >> 4, n = lane & 15;
    int K = g * 16 + e;
    int col = t * 16 + n;
    w1f[i] = (K < NB_) ? f2bf(w1[K * H_ + col]) : (unsigned short)0;
  }
  // layer 2: 9 N-tiles x 2 K-chunks (K=64 total), 144 cols
  for (int i = tid; i < 18 * 32 * 16; i += stride) {
    int e = i & 15, lane = (i >> 4) & 31, f = i >> 9;
    int t = f >> 1, kc = f & 1;
    int g = lane >> 4, n = lane & 15;
    int K = kc * 32 + g * 16 + e;
    int col = t * 16 + n;
    w2f[i] = f2bf(w2[K * OC_ + col]);
  }
}

__global__ void zero_agg(float* __restrict__ agg, size_t total) {
  size_t i = (size_t)blockIdx.x * blockDim.x + threadIdx.x;
  size_t stride = (size_t)gridDim.x * blockDim.x;
  for (; i < total; i += stride) agg[i] = 0.0f;
}

// ---------------------------------------------------------------------------
// Main edge kernel: 256 threads = 8 waves, 256 edges per block.
// ---------------------------------------------------------------------------
__global__ __launch_bounds__(256) void edge_kernel(
    const int*   __restrict__ zarr,
    const float* __restrict__ pos,
    const int*   __restrict__ eidx,     // [2, E] row-major
    const float* __restrict__ shift,
    const float* __restrict__ nembw,    // [NSP, 16]
    const float* __restrict__ nembb,    // [16]
    const float* __restrict__ b1,       // [64]
    const float* __restrict__ b2,       // [144]
    const unsigned short* __restrict__ w1f,
    const unsigned short* __restrict__ w2f,
    float* __restrict__ agg,            // [N, 144]
    int E) {
  // 32B record per edge: 8 bf16 features + 8 zeros, matching the 16-bit
  // A-operand element order for g==0 lanes (e<8 -> K=e, e>=8 -> zero).
  __shared__ __align__(16) unsigned short s_feat[256 * 16];
  __shared__ __align__(16) float          s_sh[256 * SHP_];
  __shared__ __align__(16) float          s_nf[256 * C_];
  __shared__ __align__(16) unsigned short s_zero[16];       // 32B zero region
  __shared__ int                          s_dst[256];
  __shared__ __align__(16) unsigned short s_h[8][16 * H_];  // per-wave hidden

  const int tid = threadIdx.x;
  const int e   = blockIdx.x * 256 + tid;

  if (tid == 0) {
    v4u zz = {0u, 0u, 0u, 0u};
    *(v4u*)&s_zero[0] = zz;
    *(v4u*)&s_zero[8] = zz;
  }

  // ------------------ phase 1: per-edge geometry -> LDS -------------------
  if (e < E) {
    int src = eidx[e];
    int dst = eidx[E + e];
    float ex = pos[dst * 3 + 0] - pos[src * 3 + 0] + shift[e * 3 + 0];
    float ey = pos[dst * 3 + 1] - pos[src * 3 + 1] + shift[e * 3 + 1];
    float ez = pos[dst * 3 + 2] - pos[src * 3 + 2] + shift[e * 3 + 2];
    float r2   = ex * ex + ey * ey + ez * ez + 1e-12f;
    float rinv = rsqrtf(r2);
    float r    = r2 * rinv;
    float x = ex * rinv, y = ey * rinv, z = ez * rinv;

    const float s3  = 1.7320508075688772f;
    const float s5  = 2.2360679774997896f;
    const float s15 = 3.8729833462074170f;
    v4f shA = {1.0f, s3 * x, s3 * y, s3 * z};
    v4f shB = {s15 * x * y, s15 * y * z, 0.5f * s5 * (3.0f * z * z - 1.0f),
               s15 * x * z};
    *(v4f*)&s_sh[tid * SHP_ + 0] = shA;
    *(v4f*)&s_sh[tid * SHP_ + 4] = shB;
    s_sh[tid * SHP_ + 8] = 0.5f * s15 * (x * x - y * y);

    float t    = r * (1.0f / RCUT_);
    float env  = (t < 1.0f) ? 0.5f * (__cosf(PI_ * t) + 1.0f) : 0.0f;
    float pref = 0.63245553203367586f /* sqrt(2/RCUT) */ * rinv * env;
    float warg = PI_ * r * (1.0f / RCUT_);
    float bv[NB_];
#pragma unroll
    for (int n = 1; n <= NB_; ++n) bv[n - 1] = pref * __sinf((float)n * warg);
    v4u fa = {pkbf(bv[0], bv[1]), pkbf(bv[2], bv[3]),
              pkbf(bv[4], bv[5]), pkbf(bv[6], bv[7])};
    v4u zz = {0u, 0u, 0u, 0u};
    *(v4u*)&s_feat[tid * 16 + 0] = fa;
    *(v4u*)&s_feat[tid * 16 + 8] = zz;

    int zi = zarr[src];
#pragma unroll
    for (int k = 0; k < 4; ++k) {
      v4f nf = {nembw[zi * C_ + 4 * k + 0] + nembb[4 * k + 0],
                nembw[zi * C_ + 4 * k + 1] + nembb[4 * k + 1],
                nembw[zi * C_ + 4 * k + 2] + nembb[4 * k + 2],
                nembw[zi * C_ + 4 * k + 3] + nembb[4 * k + 3]};
      *(v4f*)&s_nf[tid * C_ + 4 * k] = nf;
    }
    s_dst[tid] = dst;
  } else {
    v4u zz = {0u, 0u, 0u, 0u};
    v4f zf = {0.f, 0.f, 0.f, 0.f};
    *(v4u*)&s_feat[tid * 16 + 0] = zz;
    *(v4u*)&s_feat[tid * 16 + 8] = zz;
    *(v4f*)&s_sh[tid * SHP_ + 0] = zf;
    *(v4f*)&s_sh[tid * SHP_ + 4] = zf;
    s_sh[tid * SHP_ + 8] = 0.f;
#pragma unroll
    for (int k = 0; k < 4; ++k) *(v4f*)&s_nf[tid * C_ + 4 * k] = zf;
    s_dst[tid] = -1;
  }
  __syncthreads();

  // ------------------ phase 2: WMMA MLP + scatter -------------------------
  const v16bf* w1v = (const v16bf*)w1f;
  const v16bf* w2v = (const v16bf*)w2f;
  const int w = tid >> 5, lane = tid & 31, g = lane >> 4, lm = lane & 15;

  // biases (register-resident, reused by both sub-tiles)
  float b1v[4], b2v[NSH_];
#pragma unroll
  for (int t = 0; t < 4; ++t) b1v[t] = b1[t * 16 + lm];
#pragma unroll
  for (int t = 0; t < NSH_; ++t) b2v[t] = b2[t * 16 + lm];

  // layer-1 B fragments, resident (4 x 32B = 32 VGPRs)
  v16bf w1b[4];
#pragma unroll
  for (int t = 0; t < 4; ++t) w1b[t] = w1v[t * 32 + lane];

  for (int sub = 0; sub < 2; ++sub) {
    const int tb = (w << 5) + (sub << 4);    // 16-edge tile base in block

    // ---- layer 1 A: one 32B LDS load, zero region for g==1 lanes ----
    AFrag a1;
    const v16bf* asrc = (g == 0) ? (const v16bf*)&s_feat[(tb + lm) * 16]
                                 : (const v16bf*)&s_zero[0];
    a1.v = *asrc;

    v8f h[4];
#pragma unroll
    for (int t = 0; t < 4; ++t) {
      v8f c = {0.f, 0.f, 0.f, 0.f, 0.f, 0.f, 0.f, 0.f};
      c = __builtin_amdgcn_wmma_f32_16x16x32_bf16(false, a1.v, false, w1b[t],
                                                  (short)0, c, false, false);
      h[t] = c;
    }

    // ---- bias + SiLU (v_exp + v_rcp), re-stripe via per-wave LDS ----
#pragma unroll
    for (int t = 0; t < 4; ++t) {
#pragma unroll
      for (int r = 0; r < 8; ++r) {
        float xv = h[t][r] + b1v[t];
        float sv = xv * __builtin_amdgcn_rcpf(1.0f + __expf(-xv));
        s_h[w][(r + 8 * g) * H_ + t * 16 + lm] = f2bf(sv);
      }
    }
    // same-wave LDS ops are in-order: no barrier needed before re-read

    // ---- layer 2 A fragments: 2 chunks, 2 x b128 each, built ONCE ----
    AFrag a2[2];
    {
      const char* hrow = (const char*)&s_h[w][lm * H_];  // 128B row
#pragma unroll
      for (int kc = 0; kc < 2; ++kc) {
        a2[kc].q[0] = *(const v4u*)(hrow + kc * 64 + g * 32);
        a2[kc].q[1] = *(const v4u*)(hrow + kc * 64 + g * 32 + 16);
      }
    }

    // ---- layer 2: 9 N-tiles, all accumulators live ----
    v8f acc[NSH_];
#pragma unroll
    for (int t = 0; t < NSH_; ++t) {
      v16bf bb0 = w2v[(t * 2 + 0) * 32 + lane];
      v8f c = {0.f, 0.f, 0.f, 0.f, 0.f, 0.f, 0.f, 0.f};
      c = __builtin_amdgcn_wmma_f32_16x16x32_bf16(false, a2[0].v, false, bb0,
                                                  (short)0, c, false, false);
      v16bf bb1 = w2v[(t * 2 + 1) * 32 + lane];
      c = __builtin_amdgcn_wmma_f32_16x16x32_bf16(false, a2[1].v, false, bb1,
                                                  (short)0, c, false, false);
      acc[t] = c;
    }

    // ---- scale by sh * node_feat and atomic-scatter (r outer) ----
#pragma unroll
    for (int r = 0; r < 8; ++r) {
      int m = tb + r + 8 * g;                 // edge row in block
      int d = s_dst[m];
      float nf = s_nf[m * C_ + lm];
      v4f sA = *(const v4f*)&s_sh[m * SHP_ + 0];
      v4f sB = *(const v4f*)&s_sh[m * SHP_ + 4];
      float s8 = s_sh[m * SHP_ + 8];
      float shv[NSH_] = {sA.x, sA.y, sA.z, sA.w, sB.x, sB.y, sB.z, sB.w, s8};
      if (d >= 0) {
        float* base = agg + (size_t)d * OC_ + lm;
#pragma unroll
        for (int t = 0; t < NSH_; ++t) {
          atomicAdd(base + t * 16, (acc[t][r] + b2v[t]) * shv[t] * nf);
        }
      }
    }
  }
}

// ---------------------------------------------------------------------------
// forces = (agg / sqrt(16)) @ out_w + out_b
// ---------------------------------------------------------------------------
__global__ void force_kernel(const float* __restrict__ agg,
                             const float* __restrict__ ow,
                             const float* __restrict__ ob,
                             float* __restrict__ out, int N) {
  int n = blockIdx.x * blockDim.x + threadIdx.x;
  if (n >= N) return;
  float o0 = ob[0], o1 = ob[1], o2 = ob[2];
  const float scale = 0.25f;                       // 1/sqrt(AVG_DEG=16)
  const v4f* row = (const v4f*)(agg + (size_t)n * OC_);   // 144*4B, 16B rows
#pragma unroll 4
  for (int j4 = 0; j4 < OC_ / 4; ++j4) {
    v4f a = row[j4];
#pragma unroll
    for (int k = 0; k < 4; ++k) {
      int j = j4 * 4 + k;
      float av = ((const float*)&a)[k] * scale;
      o0 += av * ow[j * 3 + 0];
      o1 += av * ow[j * 3 + 1];
      o2 += av * ow[j * 3 + 2];
    }
  }
  out[n * 3 + 0] = o0;
  out[n * 3 + 1] = o1;
  out[n * 3 + 2] = o2;
}

extern "C" void kernel_launch(void* const* d_in, const int* in_sizes, int n_in,
                              void* d_out, int out_size, void* d_ws, size_t ws_size,
                              hipStream_t stream) {
  const int*   z     = (const int*)  d_in[0];
  const float* pos   = (const float*)d_in[1];
  const int*   eidx  = (const int*)  d_in[2];
  const float* shift = (const float*)d_in[3];
  const float* nembw = (const float*)d_in[4];
  const float* nembb = (const float*)d_in[5];
  const float* w1    = (const float*)d_in[6];
  const float* b1    = (const float*)d_in[7];
  const float* w2    = (const float*)d_in[8];
  const float* b2    = (const float*)d_in[9];
  const float* ow    = (const float*)d_in[10];
  const float* ob    = (const float*)d_in[11];
  float* out = (float*)d_out;

  const int N = in_sizes[0];          // nodes
  const int E = in_sizes[2] / 2;      // edges (edge_index is [2, E])

  float* agg = (float*)d_ws;
  unsigned short* w1f =
      (unsigned short*)((char*)d_ws + (size_t)N * OC_ * sizeof(float));
  unsigned short* w2f = w1f + 4 * 32 * 16;

  // 1) pack weights into WMMA B-operand fragment layout
  prep_frags<<<44, 256, 0, stream>>>(w1, w2, w1f, w2f);

  // 2) zero the segment-sum accumulator (L2-resident, 28.8 MB)
  size_t aggN = (size_t)N * OC_;
  int zb = (int)((aggN + 255) / 256);
  zero_agg<<<zb, 256, 0, stream>>>(agg, aggN);

  // 3) per-edge geometry + WMMA MLP + atomic scatter
  int eb = (E + 255) / 256;
  edge_kernel<<<eb, 256, 0, stream>>>(z, pos, eidx, shift, nembw, nembb,
                                      b1, b2, w1f, w2f, agg, E);

  // 4) epilogue GEMV: forces = agg/4 @ out_w + out_b
  int nb = (N + 255) / 256;
  force_kernel<<<nb, 256, 0, stream>>>(agg, ow, ob, out, N);
}
